// ConstraintGNN_55843164782680
// MI455X (gfx1250) — matrix operand order repaired and checked by reading, not compiled, not run on hardware
//
#include <hip/hip_runtime.h>

typedef __attribute__((ext_vector_type(16))) _Float16 v16h;
typedef __attribute__((ext_vector_type(8)))  float    v8f;

#define NFACTS 200000
#define NCONS  50000
#define NEDGES 3200000
#define HID    64

// ---------------------------------------------------------------------------
// WMMA fragment loaders (LDS, f16).
// A (16-bit, 16x32 per step): lane's fragment = one contiguous 32B run:
//   K in [kb*32 + grp*16, +16), row M = lane%16.
// B from transposed weights Wt[n][k]: lane's fragment = one contiguous 32B run:
//   N = nt*16 + lane%16, K in [kb*32 + grp*16, +16).
// ---------------------------------------------------------------------------
__device__ __forceinline__ v16h load_a_frag(const _Float16* As, int sA, int kb, int lane) {
    int grp = (lane >> 4) & 1, mr = lane & 15;
    const _Float16* ap = As + mr * sA + kb * 32 + 16 * grp;
    v16h a;
#pragma unroll
    for (int i = 0; i < 16; ++i) a[i] = ap[i];
    return a;
}

__device__ __forceinline__ v16h load_b_frag(const _Float16* Bt, int sB, int kb, int nt, int lane) {
    int grp = (lane >> 4) & 1, mr = lane & 15;
    const _Float16* bp = Bt + (nt * 16 + mr) * sB + kb * 32 + grp * 16;
    v16h b;
#pragma unroll
    for (int i = 0; i < 16; ++i) b[i] = bp[i];
    return b;
}

#define WMMA_F16(a, b, c) \
    __builtin_amdgcn_wmma_f32_16x16x32_f16(false, (a), false, (b), (short)0, (c), false, false)

// ---------------------------------------------------------------------------
// Zero accumulators (vectorized b128 stores).
// ---------------------------------------------------------------------------
__global__ void zero_kernel(float4* __restrict__ p, int n4) {
    int i = blockIdx.x * blockDim.x + threadIdx.x;
    int stride = gridDim.x * blockDim.x;
    float4 z = make_float4(0.f, 0.f, 0.f, 0.f);
    for (; i < n4; i += stride) p[i] = z;
}

// ---------------------------------------------------------------------------
// Encoder: H = relu(X @ W1 + b1) @ W2 + b2.  Layer 1 (K=5/10) on VALU,
// layer 2 via WMMA (kb outer / nt inner: 4 independent accumulators).
// One wave = one 16-row tile; 8 tiles per 256-thread block.
// ---------------------------------------------------------------------------
template <int KIN>
__global__ __launch_bounds__(256) void encoder_kernel(
    const float* __restrict__ X, const float* __restrict__ W1,
    const float* __restrict__ b1, const float* __restrict__ W2,
    const float* __restrict__ b2, float* __restrict__ H, int ntiles) {
    __shared__ _Float16 Wt[64 * 64];       // W2 transposed, f16 (8 KB)
    __shared__ float    W1s[KIN * 64];
    __shared__ float    b1s[64], b2s[64];
    __shared__ float    Xs[8][16 * KIN];
    __shared__ _Float16 As[8][16 * 64];

    int tid = threadIdx.x;
    for (int i = tid; i < 64 * 64; i += 256) {
        int n = i >> 6, k = i & 63;
        Wt[i] = (_Float16)W2[k * 64 + n];
    }
    for (int i = tid; i < KIN * 64; i += 256) W1s[i] = W1[i];
    if (tid < 64) { b1s[tid] = b1[tid]; b2s[tid] = b2[tid]; }
    __syncthreads();

    int wave = tid >> 5, lane = tid & 31;
    int tile = blockIdx.x * 8 + wave;
    if (tile >= ntiles) return;          // wave-uniform: EXEC all-ones below
    int row0 = tile * 16;

    for (int i = lane; i < 16 * KIN; i += 32) Xs[wave][i] = X[row0 * KIN + i];

    // layer 1: 16x64 outputs, K=KIN, VALU; relu result becomes f16 A tile
    for (int nn = lane; nn < 64; nn += 32) {
#pragma unroll
        for (int m = 0; m < 16; ++m) {
            float acc = b1s[nn];
#pragma unroll
            for (int k = 0; k < KIN; ++k)
                acc += Xs[wave][m * KIN + k] * W1s[k * 64 + nn];
            As[wave][m * 64 + nn] = (_Float16)fmaxf(acc, 0.0f);
        }
    }

    // layer 2: A[16x64] @ W2[64x64]; 2 K-steps, 4 interleaved accumulators
    v8f c[4] = {};
#pragma unroll
    for (int kb = 0; kb < 2; ++kb) {
        v16h a = load_a_frag(As[wave], 64, kb, lane);
#pragma unroll
        for (int nt = 0; nt < 4; ++nt)
            c[nt] = WMMA_F16(a, load_b_frag(Wt, 64, kb, nt, lane), c[nt]);
    }
    int grp = lane >> 4, mr = lane & 15;
#pragma unroll
    for (int nt = 0; nt < 4; ++nt) {
        int n = nt * 16 + mr;
#pragma unroll
        for (int r = 0; r < 8; ++r)
            H[(row0 + r + 8 * grp) * HID + n] = c[nt][r] + b2s[n];
    }
}

// ---------------------------------------------------------------------------
// Edge scatter: one wave per edge, 2 floats per lane.  constraint_h (12.8 MB)
// and summed (51 MB) stay L2-resident; relaxed agent-scope f32 atomics resolve
// in L2's atomic units.  Indices are the only HBM stream -> prefetch them.
// ---------------------------------------------------------------------------
__global__ __launch_bounds__(256) void edge_kernel(
    const int* __restrict__ edges, const float* __restrict__ ch,
    float* __restrict__ summed, float* __restrict__ cnt, int nedges) {
    int lane = threadIdx.x & 31;
    int wave = (blockIdx.x * blockDim.x + threadIdx.x) >> 5;
    int nwaves = (gridDim.x * blockDim.x) >> 5;
    for (int e = wave; e < nedges; e += nwaves) {
        int ep = e + 8 * nwaves;
        if (ep < nedges) {
            __builtin_prefetch(&edges[ep], 0, 1);           // global_prefetch_b8
            __builtin_prefetch(&edges[nedges + ep], 0, 1);
        }
        int src = __builtin_amdgcn_readfirstlane(edges[e]);
        int dst = __builtin_amdgcn_readfirstlane(edges[nedges + e]);
        const float2 v = ((const float2*)(ch + (size_t)dst * HID))[lane];
        float* srow = summed + (size_t)src * HID + lane * 2;
        __hip_atomic_fetch_add(&srow[0], v.x, __ATOMIC_RELAXED, __HIP_MEMORY_SCOPE_AGENT);
        __hip_atomic_fetch_add(&srow[1], v.y, __ATOMIC_RELAXED, __HIP_MEMORY_SCOPE_AGENT);
        if (lane == 0)
            __hip_atomic_fetch_add(&cnt[src], 1.0f, __ATOMIC_RELAXED, __HIP_MEMORY_SCOPE_AGENT);
    }
}

// ---------------------------------------------------------------------------
// Merge: fh = (cnt>0) ? concat([fh, summed/max(cnt,1)]) @ W1 + b1 : fh
// A is 16x128 (concat), W1:[128x64]; 4 K-steps x 4 interleaved accumulators.
// Branch-free select at store (unconditional load + v_cndmask).  In place.
// ---------------------------------------------------------------------------
__global__ __launch_bounds__(256) void merge_kernel(
    float* __restrict__ fh, const float* __restrict__ summed,
    const float* __restrict__ cnt, const float* __restrict__ W1,
    const float* __restrict__ b1, int ntiles) {
    __shared__ _Float16 Wt[64 * 128];      // W1 transposed, f16 (16 KB)
    __shared__ float    b1s[64];
    __shared__ _Float16 As[8][16 * 128];   // concat tile per wave (32 KB)

    int tid = threadIdx.x;
    for (int i = tid; i < 64 * 128; i += 256) {
        int n = i >> 7, k = i & 127;
        Wt[i] = (_Float16)W1[k * 64 + n];
    }
    if (tid < 64) b1s[tid] = b1[tid];
    __syncthreads();

    int wave = tid >> 5, lane = tid & 31;
    int tile = blockIdx.x * 8 + wave;
    if (tile >= ntiles) return;
    int row0 = tile * 16;

    for (int i = lane; i < 16 * 64; i += 32) {
        int m = i >> 6, n = i & 63;
        As[wave][m * 128 + n] = (_Float16)fh[(row0 + m) * HID + n];
        float cv = cnt[row0 + m];
        float mean = summed[(row0 + m) * HID + n] / fmaxf(cv, 1.0f);
        As[wave][m * 128 + 64 + n] = (_Float16)mean;
    }

    v8f c[4] = {};
#pragma unroll
    for (int kb = 0; kb < 4; ++kb) {
        v16h a = load_a_frag(As[wave], 128, kb, lane);
#pragma unroll
        for (int nt = 0; nt < 4; ++nt)
            c[nt] = WMMA_F16(a, load_b_frag(Wt, 128, kb, nt, lane), c[nt]);
    }

    int grp = lane >> 4, mr = lane & 15;
#pragma unroll
    for (int nt = 0; nt < 4; ++nt) {
        int n = nt * 16 + mr;
#pragma unroll
        for (int r = 0; r < 8; ++r) {
            int m = r + 8 * grp;
            int idx = (row0 + m) * HID + n;
            float old = fh[idx];                       // unconditional load
            float upd = c[nt][r] + b1s[n];
            fh[idx] = (cnt[row0 + m] > 0.0f) ? upd : old;  // v_cndmask
        }
    }
}

// ---------------------------------------------------------------------------
// Head: h = relu(fh @ W2 + b2); s = relu(h @ Ws1 + bs1); out = s @ Ws2 + bs2
// Two WMMA stages (interleaved accumulators), final 32-dot on VALU.
// ---------------------------------------------------------------------------
__global__ __launch_bounds__(256) void head_kernel(
    const float* __restrict__ fh, const float* __restrict__ W2,
    const float* __restrict__ b2, const float* __restrict__ Ws1,
    const float* __restrict__ bs1, const float* __restrict__ Ws2,
    const float* __restrict__ bs2, float* __restrict__ out, int ntiles) {
    __shared__ _Float16 W2t[64 * 64];      // 8 KB
    __shared__ _Float16 Ws1t[32 * 64];     // 4 KB
    __shared__ float    b2s[64], bs1s[32], ws2s[32];
    __shared__ float    bs2s;
    __shared__ _Float16 As[8][16 * 64];    // 16 KB
    __shared__ float    Ss[8][16 * 32];    // 16 KB

    int tid = threadIdx.x;
    for (int i = tid; i < 64 * 64; i += 256) {
        int n = i >> 6, k = i & 63;
        W2t[i] = (_Float16)W2[k * 64 + n];
    }
    for (int i = tid; i < 32 * 64; i += 256) {
        int n = i >> 6, k = i & 63;
        Ws1t[i] = (_Float16)Ws1[k * 32 + n];
    }
    if (tid < 64) b2s[tid] = b2[tid];
    if (tid < 32) { bs1s[tid] = bs1[tid]; ws2s[tid] = Ws2[tid]; }
    if (tid == 0) bs2s = bs2[0];
    __syncthreads();

    int wave = tid >> 5, lane = tid & 31;
    int tile = blockIdx.x * 8 + wave;
    if (tile >= ntiles) return;
    int row0 = tile * 16;
    int grp = lane >> 4, mr = lane & 15;

    for (int i = lane; i < 16 * 64; i += 32)
        As[wave][i] = (_Float16)fh[row0 * HID + i];

    // stage 1: h = relu(fh @ W2 + b2) -> rewrite As (f16)
    v8f hc[4] = {};
#pragma unroll
    for (int kb = 0; kb < 2; ++kb) {
        v16h a = load_a_frag(As[wave], 64, kb, lane);
#pragma unroll
        for (int nt = 0; nt < 4; ++nt)
            hc[nt] = WMMA_F16(a, load_b_frag(W2t, 64, kb, nt, lane), hc[nt]);
    }
#pragma unroll
    for (int nt = 0; nt < 4; ++nt) {
        int n = nt * 16 + mr;
#pragma unroll
        for (int r = 0; r < 8; ++r)
            As[wave][(r + 8 * grp) * 64 + n] = (_Float16)fmaxf(hc[nt][r] + b2s[n], 0.0f);
    }

    // stage 2: s = relu(h @ Ws1 + bs1) -> Ss (16x32, f32)
    v8f sc[2] = {};
#pragma unroll
    for (int kb = 0; kb < 2; ++kb) {
        v16h a = load_a_frag(As[wave], 64, kb, lane);
#pragma unroll
        for (int nt = 0; nt < 2; ++nt)
            sc[nt] = WMMA_F16(a, load_b_frag(Ws1t, 64, kb, nt, lane), sc[nt]);
    }
#pragma unroll
    for (int nt = 0; nt < 2; ++nt) {
        int n = nt * 16 + mr;
#pragma unroll
        for (int r = 0; r < 8; ++r)
            Ss[wave][(r + 8 * grp) * 32 + n] = fmaxf(sc[nt][r] + bs1s[n], 0.0f);
    }

    // stage 3: out = s @ Ws2 + bs2 (length-32 dot, one row per lane)
    if (lane < 16) {
        float acc = bs2s;
#pragma unroll
        for (int k = 0; k < 32; ++k) acc += Ss[wave][lane * 32 + k] * ws2s[k];
        out[row0 + lane] = acc;
    }
}

// ---------------------------------------------------------------------------
extern "C" void kernel_launch(void* const* d_in, const int* in_sizes, int n_in,
                              void* d_out, int out_size, void* d_ws, size_t ws_size,
                              hipStream_t stream) {
    (void)in_sizes; (void)n_in; (void)out_size; (void)ws_size;

    const float* fact_features       = (const float*)d_in[0];
    const float* constraint_features = (const float*)d_in[1];
    const int*   edges               = (const int*)d_in[2];
    const float* Wf1 = (const float*)d_in[3];
    const float* bf1 = (const float*)d_in[4];
    const float* Wf2 = (const float*)d_in[5];
    const float* bf2 = (const float*)d_in[6];
    const float* Wc1 = (const float*)d_in[7];
    const float* bc1 = (const float*)d_in[8];
    const float* Wc2 = (const float*)d_in[9];
    const float* bc2 = (const float*)d_in[10];
    const float* W1  = (const float*)d_in[11];
    const float* b1  = (const float*)d_in[12];
    const float* W2  = (const float*)d_in[13];
    const float* b2  = (const float*)d_in[14];
    const float* Ws1 = (const float*)d_in[15];
    const float* bs1 = (const float*)d_in[16];
    const float* Ws2 = (const float*)d_in[17];
    const float* bs2 = (const float*)d_in[18];
    float* out = (float*)d_out;

    // workspace layout (floats): fact_h | constraint_h | summed | cnt
    float* ws           = (float*)d_ws;
    float* fact_h       = ws;
    float* constraint_h = fact_h + (size_t)NFACTS * HID;
    float* summed       = constraint_h + (size_t)NCONS * HID;
    float* cnt          = summed + (size_t)NFACTS * HID;

    const int fact_tiles = NFACTS / 16;  // 12500 (exact)
    const int cons_tiles = NCONS / 16;   //  3125 (exact)

    // 1) zero accumulators (summed + cnt contiguous, 13.0M floats = 3.25M float4)
    zero_kernel<<<2048, 256, 0, stream>>>((float4*)summed,
                                          (NFACTS * HID + NFACTS) / 4);

    // 2) encoders (WMMA)
    encoder_kernel<5><<<(cons_tiles + 7) / 8, 256, 0, stream>>>(
        constraint_features, Wc1, bc1, Wc2, bc2, constraint_h, cons_tiles);
    encoder_kernel<10><<<(fact_tiles + 7) / 8, 256, 0, stream>>>(
        fact_features, Wf1, bf1, Wf2, bf2, fact_h, fact_tiles);

    // 3) edge segment-sum (L2-resident gather + agent-scope f32 atomics)
    edge_kernel<<<8192, 256, 0, stream>>>(edges, constraint_h, summed, cnt, NEDGES);

    // 4) concat-merge GEMM (WMMA, in place)
    merge_kernel<<<(fact_tiles + 7) / 8, 256, 0, stream>>>(
        fact_h, summed, cnt, W1, b1, fact_tiles);

    // 5) MLP head (WMMA x2) -> scores
    head_kernel<<<(fact_tiles + 7) / 8, 256, 0, stream>>>(
        fact_h, W2, b2, Ws1, bs1, Ws2, bs2, out, fact_tiles);
}
